// LSTM_29515015258161
// MI455X (gfx1250) — matrix operand order repaired and checked by reading, not compiled, not run on hardware
//
#include <hip/hip_runtime.h>
#include <hip/hip_bf16.h>
#include <stdint.h>

// ---------------- CDNA5 WMMA types ----------------
typedef __attribute__((ext_vector_type(16))) __bf16 v16bf;
typedef __attribute__((ext_vector_type(8)))  float  v8f;

#define T_   256
#define B_   128
#define D_   256
#define H_   1024
#define C_   1000
#define KTOT 1280   // D_ + H_ (concatenated [x_t | h_{t-1}])

#define BM   32     // block M tile (2 wave rows)
#define BN   64     // block N tile per gate (4 wave cols)
#define BK   32     // K per bf16 WMMA

union Frag { v16bf v; uint4 q[2]; };

// ---------------- helpers ----------------
__device__ __forceinline__ uint16_t f2bf(float f) {
  union { float f; uint32_t u; } x; x.f = f;
  uint32_t u = x.u;
  if ((u & 0x7F800000u) == 0x7F800000u)            // Inf / NaN
    return (uint16_t)((u >> 16) | ((u & 0xFFFFu) ? 0x40u : 0u));
  return (uint16_t)((u + 0x7FFFu + ((u >> 16) & 1u)) >> 16);
}

__device__ __forceinline__ float sigmoidf_(float x) {
  return 1.0f / (1.0f + __expf(-x));
}

// ---------------- one-time prep kernels ----------------

// Stacked, N-major (B^T) bf16 weights: Wcat[g][n][k], k<256 from W*x, else W*h.
__global__ void prep_wcat(const float* __restrict__ Wgx, const float* __restrict__ Wgh,
                          const float* __restrict__ Wix, const float* __restrict__ Wih,
                          const float* __restrict__ Wfx, const float* __restrict__ Wfh,
                          const float* __restrict__ Wox, const float* __restrict__ Woh,
                          uint16_t* __restrict__ Wcat) {
  int idx = blockIdx.x * 256 + threadIdx.x;        // 4*1024*1280 = 5,242,880
  int k = idx % KTOT;
  int r = idx / KTOT;
  int n = r % H_;
  int g = r / H_;
  const float* Wx[4] = {Wgx, Wix, Wfx, Wox};
  const float* Wh[4] = {Wgh, Wih, Wfh, Woh};
  float v = (k < D_) ? Wx[g][k * H_ + n] : Wh[g][(k - D_) * H_ + n];
  Wcat[idx] = f2bf(v);
}

// N-major bf16 Wph, N padded 1000 -> 1024.
__global__ void prep_wph(const float* __restrict__ Wph, uint16_t* __restrict__ WphT) {
  int idx = blockIdx.x * 256 + threadIdx.x;        // 1024*1024
  int k = idx % H_;
  int n = idx / H_;
  float v = (n < C_) ? Wph[k * C_ + n] : 0.0f;
  WphT[idx] = f2bf(v);
}

__global__ void prep_x(const float* __restrict__ x, uint16_t* __restrict__ xbf) {
  int idx = blockIdx.x * 256 + threadIdx.x;        // 256*128*256 = 8,388,608
  xbf[idx] = f2bf(x[idx]);
}

// Zero h ping-pong buffers + c state, build stacked bias.
__global__ void prep_state(const float* __restrict__ bg, const float* __restrict__ bi,
                           const float* __restrict__ bf, const float* __restrict__ bo,
                           float* __restrict__ bcat, uint16_t* __restrict__ hbuf,
                           float* __restrict__ cst) {
  int idx = blockIdx.x * 256 + threadIdx.x;        // 262,144
  hbuf[idx] = 0;                                   // both h buffers
  if (idx < B_ * H_) cst[idx] = 0.0f;
  if (idx < 4 * H_) {
    const float* bs[4] = {bg, bi, bf, bo};
    bcat[idx] = bs[idx >> 10][idx & (H_ - 1)];
  }
}

// ---------------- fused LSTM step: 4-gate GEMM + cell update ----------------
// No LDS: fragments are contiguous in memory by construction (N-major weights,
// row-major activations), so each fragment = two global_load_b128.
__global__ __launch_bounds__(256) void lstm_step(
    const uint16_t* __restrict__ xt,     // [B][D] bf16 for this timestep
    const uint16_t* __restrict__ hprev,  // [B][H] bf16
    const uint16_t* __restrict__ Wcat,   // [4][H][KTOT] bf16 (N-major)
    const float*    __restrict__ bcat,   // [4][H]
    float*          __restrict__ cst,    // [B][H] fp32, tile-owned update
    uint16_t*       __restrict__ hnext)  // [B][H] bf16
{
  const int tid  = threadIdx.x;
  const int lane = tid & 31;
  const int wave = tid >> 5;
  const int wm   = wave >> 2;            // 0..1
  const int wn   = wave & 3;             // 0..3
  const bool hi  = lane >= 16;
  const int l16  = lane & 15;

  const int m0   = blockIdx.y * BM;
  const int n0   = blockIdx.x * BN;
  const int mrow = m0 + wm * 16 + l16;   // A row this lane reads
  const int ncol = n0 + wn * 16 + l16;   // B row (N index) this lane reads

  // Fragment chunk offsets (bf16 elements) within the 32-wide K slab.
  const int a0 = hi ? 16 : 0;            // A chunk0: K 0..7 / 16..23
  const int a1 = hi ? 24 : 8;            // A chunk1: K 8..15 / 24..31
  const int b0 = hi ? 16 : 0;            // B chunk:  K 0..15 / 16..31

  const uint16_t* ax  = xt    + mrow * D_;
  const uint16_t* ah  = hprev + mrow * H_;
  const uint16_t* pb0 = Wcat + (size_t)(0 * H_ + ncol) * KTOT + b0;
  const uint16_t* pb1 = Wcat + (size_t)(1 * H_ + ncol) * KTOT + b0;
  const uint16_t* pb2 = Wcat + (size_t)(2 * H_ + ncol) * KTOT + b0;
  const uint16_t* pb3 = Wcat + (size_t)(3 * H_ + ncol) * KTOT + b0;

  v8f zero = {0.f, 0.f, 0.f, 0.f, 0.f, 0.f, 0.f, 0.f};
  v8f acc[4] = {zero, zero, zero, zero};

  // ---- x region: K = 0..255
#pragma unroll 2
  for (int k0 = 0; k0 < D_; k0 += BK) {
    Frag af;
    af.q[0] = *(const uint4*)(ax + k0 + a0);
    af.q[1] = *(const uint4*)(ax + k0 + a1);
    const uint16_t* pb[4] = {pb0, pb1, pb2, pb3};
#pragma unroll
    for (int g = 0; g < 4; ++g) {
      Frag bfr;
      bfr.q[0] = *(const uint4*)(pb[g] + k0);
      bfr.q[1] = *(const uint4*)(pb[g] + k0 + 8);
      acc[g] = __builtin_amdgcn_wmma_f32_16x16x32_bf16(
          false, af.v, false, bfr.v, (short)0, acc[g], false, false);
    }
  }

  // ---- h region: K = 256..1279
#pragma unroll 2
  for (int k0 = 0; k0 < H_; k0 += BK) {
    Frag af;
    af.q[0] = *(const uint4*)(ah + k0 + a0);
    af.q[1] = *(const uint4*)(ah + k0 + a1);
    __builtin_prefetch(pb0 + D_ + k0 + 2 * BK, 0, 0);   // weight-stream prefetch
    const uint16_t* pb[4] = {pb0, pb1, pb2, pb3};
#pragma unroll
    for (int g = 0; g < 4; ++g) {
      Frag bfr;
      bfr.q[0] = *(const uint4*)(pb[g] + D_ + k0);
      bfr.q[1] = *(const uint4*)(pb[g] + D_ + k0 + 8);
      acc[g] = __builtin_amdgcn_wmma_f32_16x16x32_bf16(
          false, af.v, false, bfr.v, (short)0, acc[g], false, false);
    }
  }

  // ---- fused epilogue: bias + nonlinearities + cell/hidden update
  const float bgv = bcat[0 * H_ + ncol];
  const float biv = bcat[1 * H_ + ncol];
  const float bfv = bcat[2 * H_ + ncol];
  const float bov = bcat[3 * H_ + ncol];
#pragma unroll
  for (int j = 0; j < 8; ++j) {
    int mr = m0 + wm * 16 + j + (hi ? 8 : 0);
    float g = tanhf(acc[0][j] + bgv);
    float i = sigmoidf_(acc[1][j] + biv);
    float f = sigmoidf_(acc[2][j] + bfv);
    float o = sigmoidf_(acc[3][j] + bov);
    float cold = cst[mr * H_ + ncol];
    float cnew = g * i + cold * f;
    cst[mr * H_ + ncol] = cnew;
    float hv = tanhf(cnew) * o;
    hnext[mr * H_ + ncol] = f2bf(hv);
  }
}

// ---------------- final projection: out = h @ Wph + bp ----------------
__global__ __launch_bounds__(256) void final_proj(
    const uint16_t* __restrict__ hfin,   // [B][H] bf16
    const uint16_t* __restrict__ WphT,   // [1024][1024] bf16 (N-major, padded)
    const float*    __restrict__ bp,     // [C]
    float*          __restrict__ out)    // [B][C] fp32
{
  const int tid  = threadIdx.x;
  const int lane = tid & 31;
  const int wave = tid >> 5;
  const int wm   = wave >> 2;
  const int wn   = wave & 3;
  const bool hi  = lane >= 16;
  const int l16  = lane & 15;

  const int m0   = blockIdx.y * BM;
  const int n0   = blockIdx.x * BN;
  const int mrow = m0 + wm * 16 + l16;
  const int ncol = n0 + wn * 16 + l16;

  const int a0 = hi ? 16 : 0;
  const int a1 = hi ? 24 : 8;
  const int b0 = hi ? 16 : 0;

  const uint16_t* ar = hfin + mrow * H_;
  const uint16_t* br = WphT + (size_t)ncol * H_ + b0;

  v8f acc = {0.f, 0.f, 0.f, 0.f, 0.f, 0.f, 0.f, 0.f};

#pragma unroll 2
  for (int k0 = 0; k0 < H_; k0 += BK) {
    Frag af, bfr;
    af.q[0]  = *(const uint4*)(ar + k0 + a0);
    af.q[1]  = *(const uint4*)(ar + k0 + a1);
    bfr.q[0] = *(const uint4*)(br + k0);
    bfr.q[1] = *(const uint4*)(br + k0 + 8);
    acc = __builtin_amdgcn_wmma_f32_16x16x32_bf16(
        false, af.v, false, bfr.v, (short)0, acc, false, false);
  }

  if (ncol < C_) {
    const float bpv = bp[ncol];
#pragma unroll
    for (int j = 0; j < 8; ++j) {
      int mr = m0 + wm * 16 + j + (hi ? 8 : 0);
      out[mr * C_ + ncol] = acc[j] + bpv;
    }
  }
}

// ---------------- host launcher ----------------
extern "C" void kernel_launch(void* const* d_in, const int* in_sizes, int n_in,
                              void* d_out, int out_size, void* d_ws, size_t ws_size,
                              hipStream_t stream) {
  (void)in_sizes; (void)n_in; (void)out_size; (void)ws_size;

  const float* x   = (const float*)d_in[0];
  const float* Wgx = (const float*)d_in[1];
  const float* Wgh = (const float*)d_in[2];
  const float* Wix = (const float*)d_in[3];
  const float* Wih = (const float*)d_in[4];
  const float* Wfx = (const float*)d_in[5];
  const float* Wfh = (const float*)d_in[6];
  const float* Wox = (const float*)d_in[7];
  const float* Woh = (const float*)d_in[8];
  const float* Wph = (const float*)d_in[9];
  const float* bg  = (const float*)d_in[10];
  const float* bi  = (const float*)d_in[11];
  const float* bf  = (const float*)d_in[12];
  const float* bo  = (const float*)d_in[13];
  const float* bp  = (const float*)d_in[14];

  char* ws = (char*)d_ws;
  uint16_t* Wcat = (uint16_t*)(ws);                 // 4*1024*1280*2 = 10,485,760 B
  uint16_t* WphT = (uint16_t*)(ws + 10485760);      // 1024*1024*2   =  2,097,152 B
  uint16_t* xbf  = (uint16_t*)(ws + 12582912);      // 256*128*256*2 = 16,777,216 B
  float*    bcat = (float*)   (ws + 29360128);      // 4096*4        =     16,384 B
  uint16_t* hbuf = (uint16_t*)(ws + 29376512);      // 2*128*1024*2  =    524,288 B
  float*    cst  = (float*)   (ws + 29900800);      // 128*1024*4    =    524,288 B

  prep_wcat <<<20480, 256, 0, stream>>>(Wgx, Wgh, Wix, Wih, Wfx, Wfh, Wox, Woh, Wcat);
  prep_wph  <<< 4096, 256, 0, stream>>>(Wph, WphT);
  prep_x    <<<32768, 256, 0, stream>>>(x, xbf);
  prep_state<<< 1024, 256, 0, stream>>>(bg, bi, bf, bo, bcat, hbuf, cst);

  dim3 sgrid(H_ / BN, B_ / BM);                     // (16, 4)
  for (int t = 0; t < T_; ++t) {
    const uint16_t* xt    = xbf  + (size_t)t * B_ * D_;
    const uint16_t* hprev = hbuf + (size_t)(t & 1) * B_ * H_;
    uint16_t*       hnext = hbuf + (size_t)((t + 1) & 1) * B_ * H_;
    lstm_step<<<sgrid, 256, 0, stream>>>(xt, hprev, Wcat, bcat, cst, hnext);
  }

  const uint16_t* hfin = hbuf + (size_t)(T_ & 1) * B_ * H_;  // T even -> buffer 0
  dim3 pgrid(H_ / BN, B_ / BM);                    // padded N=1024 -> 16 tiles
  final_proj<<<pgrid, 256, 0, stream>>>(hfin, WphT, bp, (float*)d_out);
}
